// GraphSAGE_42588895707554
// MI455X (gfx1250) — matrix operand order repaired and checked by reading, not compiled, not run on hardware
//
#include <hip/hip_runtime.h>
#include <hip/hip_bf16.h>

// ---------------------------------------------------------------------------
// GraphSAGE (LSTM aggregator) for MI455X / gfx1250, wave32 + WMMA f16->f32.
// All GEMMs run on v_wmma_f32_16x16x32_f16. Weights pre-swizzled into the
// per-lane B-fragment layout and staged in LDS (256KB/block); LSTM state kept
// in registers; h transposed C-layout -> A-layout through per-wave LDS.
// Round 2: burst B-fragment loads + interleaved independent accumulators so
// consecutive WMMAs never share a destination (dep distance 4), letting the
// XDL pipe stay busy instead of stalling on s_wait_dscnt + RAW chains.
// ---------------------------------------------------------------------------

typedef _Float16 h16;
typedef __attribute__((ext_vector_type(16))) _Float16 v16h;
typedef __attribute__((ext_vector_type(8)))  _Float16 v8h;
typedef __attribute__((ext_vector_type(8)))  float    v8f;

union AFrag { v16h v; v8h h[2]; };

#define N_NODES 20000
#define KNBR    16
#define DP      128                  // padded feature dim (all layers)
#define NTILES  (N_NODES / 16)       // 1250 node tiles
#define WPB     8                    // waves per block
#define LDS_WBYTES (512 * DP * 2 * 2) // Wih + Whh packed f16 = 262144 B
#define HAREA   4096                 // per-wave h transpose area [16][128] f16
#define SMEM_BYTES (LDS_WBYTES + WPB * HAREA)   // 294912 B < 320KB WGP LDS

// packed per-layer weight block offsets (bytes)
#define OFF_WIH 0
#define OFF_WHH 131072
#define OFF_WSF 262144
#define OFF_WNG 294912
#define OFF_BG  327680
#define OFF_BO  329728
#define LBLOCK  330240

// halfs per gate block in the packed Wih/Whh layout: 8 ft * 4 kc * 32 * 16
#define GSTRIDE 16384

static __device__ __forceinline__ v8f wmma_f16(v16h a, v16h b, v8f c) {
  return __builtin_amdgcn_wmma_f32_16x16x32_f16(false, a, false, b,
                                                (short)0, c, false, false);
}
static __device__ __forceinline__ float sigm(float x) {
  return 1.0f / (1.0f + __expf(-x));
}
static __device__ __forceinline__ float tanh_s(float x) {
  float a = fabsf(x);
  float e = __expf(-2.0f * a);
  float t = (1.0f - e) / (1.0f + e);
  return x < 0.0f ? -t : t;
}

// ---------------------------------------------------------------------------
// Weight pre-swizzle: pack B-operand tiles so a lane loads 16 contiguous halfs
// (= 2 x b128) per 32x16 tile. Pack linear index p:
//   i = p&15, lane = (p>>4)&31, kc = (p>>9)&3, nt = p>>11
//   K = kc*32 + 16*(lane>=16) + i ;  Nc = nt*16 + (lane&15)
// Gate blocks padded din->128, output cols padded dout->128 (zeros).
// ---------------------------------------------------------------------------
__global__ void prep_weights(const float* __restrict__ Wih,
                             const float* __restrict__ Whh,
                             const float* __restrict__ bih,
                             const float* __restrict__ bhh,
                             const float* __restrict__ Wself,
                             const float* __restrict__ Wneigh,
                             const float* __restrict__ bvec,
                             int din, int dout, char* __restrict__ dst) {
  int idx = blockIdx.x * 256 + threadIdx.x;
  h16*   wih = (h16*)(dst + OFF_WIH);
  h16*   whh = (h16*)(dst + OFF_WHH);
  h16*   wsf = (h16*)(dst + OFF_WSF);
  h16*   wng = (h16*)(dst + OFF_WNG);
  float* bg  = (float*)(dst + OFF_BG);
  float* bo  = (float*)(dst + OFF_BO);

  if (idx < 131072) {                     // Wih / Whh packs (65536 halfs each)
    const float* W = (idx < 65536) ? Wih : Whh;
    h16* out = (idx < 65536) ? wih : whh;
    int p = idx & 65535;
    int i = p & 15, lane = (p >> 4) & 31, kc = (p >> 9) & 3, nt = p >> 11;
    int k = kc * 32 + (lane >> 4) * 16 + i;
    int n = nt * 16 + (lane & 15);
    int g = n >> 7, j = n & 127;          // gate block, within-gate column
    float v = (k < din && j < din) ? W[(g * din + j) * din + k] : 0.0f;
    out[p] = (h16)v;
  } else if (idx < 163840) {              // Wself / Wneigh (16384 halfs each)
    int p2 = idx - 131072;
    const float* W = (p2 < 16384) ? Wself : Wneigh;
    h16* out = (p2 < 16384) ? wsf : wng;
    int p = p2 & 16383;
    int i = p & 15, lane = (p >> 4) & 31, kc = (p >> 9) & 3, nt = p >> 11;
    int k = kc * 32 + (lane >> 4) * 16 + i;
    int n = nt * 16 + (lane & 15);
    float v = (k < din && n < dout) ? W[n * din + k] : 0.0f;
    out[p] = (h16)v;
  } else if (idx < 164352) {              // combined gate bias (512)
    int n = idx - 163840;
    int g = n >> 7, j = n & 127;
    bg[n] = (j < din) ? (bih[g * din + j] + bhh[g * din + j]) : 0.0f;
  } else if (idx < 164480) {              // output bias (128)
    int n = idx - 164352;
    bo[n] = (n < dout) ? bvec[n] : 0.0f;
  }
}

__global__ void prep_x(const float* __restrict__ x, h16* __restrict__ h0,
                       int total) {
  int i = blockIdx.x * 256 + threadIdx.x;
  if (i < total) h0[i] = (h16)x[i];
}

// ---------------------------------------------------------------------------
// One SAGE-LSTM layer. Each wave owns a 16-node M-tile; LSTM state in regs.
// mode == 0: relu + f16 store to hout ; mode == 1: final sigmoid f32 to dfinal
// ---------------------------------------------------------------------------
extern __shared__ __align__(32) char smem[];

__global__ __launch_bounds__(256, 1)
void sage_layer(const h16* __restrict__ hin, const int* __restrict__ nbr,
                const char* __restrict__ wblk, h16* __restrict__ hout,
                float* __restrict__ dfinal, int mode) {
  const int tid  = threadIdx.x;
  const int lane = tid & 31;
  const int widx = tid >> 5;

  // Stage Wih+Whh packed weights into LDS (shared by all waves/steps).
  {
    const uint4* src = (const uint4*)(wblk + OFF_WIH);
    uint4* dst = (uint4*)smem;
    #pragma unroll
    for (int i = 0; i < LDS_WBYTES / 16 / 256; ++i)
      dst[i * 256 + tid] = src[i * 256 + tid];
  }
  __syncthreads();

  const int tile = blockIdx.x * WPB + widx;
  if (tile >= NTILES) return;             // wave-uniform; EXEC stays all-ones

  const h16* ldsWih = (const h16*)smem;
  const h16* ldsWhh = (const h16*)(smem + OFF_WHH);
  h16* hlds = (h16*)(smem + LDS_WBYTES + widx * HAREA);  // [16][128] f16

  const int row  = lane & 15;             // node row within tile (A-row / C-col)
  const int hi8  = (lane >> 4) * 8;       // 0 or 8 (half-wave select)
  const int node = tile * 16 + row;
  const float* bg = (const float*)(wblk + OFF_BG);

  v8f zz = {};
  v8h hz = {};
  v8f c[8];
  #pragma unroll
  for (int f = 0; f < 8; ++f) c[f] = zz;
  AFrag Ah[4];
  #pragma unroll
  for (int kc = 0; kc < 4; ++kc) { Ah[kc].h[0] = hz; Ah[kc].h[1] = hz; }

  #pragma unroll 1
  for (int t = 0; t < KNBR; ++t) {
    const int nb = nbr[node * KNBR + t];
    const h16* mrow = hin + (size_t)nb * DP;
    {   // prefetch next step's neighbor row (global_prefetch_b8)
      const int tn = (t + 1 < KNBR) ? t + 1 : t;
      const int nb2 = nbr[node * KNBR + tn];
      __builtin_prefetch(hin + (size_t)nb2 * DP, 0, 3);
    }
    AFrag Am[4];
    #pragma unroll
    for (int kc = 0; kc < 4; ++kc) {      // A-fragment: 2 x 16B per 32-K chunk
      Am[kc].h[0] = *(const v8h*)(mrow + kc * 32 + hi8 * 2);
      Am[kc].h[1] = *(const v8h*)(mrow + kc * 32 + hi8 * 2 + 8);
    }

    #pragma unroll
    for (int ft = 0; ft < 8; ++ft) {      // 8 feature tiles of 16 columns
      v8f G[4];
      G[0] = zz; G[1] = zz; G[2] = zz; G[3] = zz;
      #pragma unroll
      for (int kc = 0; kc < 4; ++kc) {
        const int toff = ft * 2048 + kc * 512 + lane * 16;
        // Burst-load B fragments for all 4 gates, then issue 4 WMMAs with
        // independent accumulators (no back-to-back same-dest WMMA).
        v16h Bi[4], Bh[4];
        #pragma unroll
        for (int g = 0; g < 4; ++g)
          Bi[g] = *(const v16h*)(ldsWih + g * GSTRIDE + toff);
        #pragma unroll
        for (int g = 0; g < 4; ++g)
          Bh[g] = *(const v16h*)(ldsWhh + g * GSTRIDE + toff);
        #pragma unroll
        for (int g = 0; g < 4; ++g) G[g] = wmma_f16(Am[kc].v, Bi[g], G[g]);
        #pragma unroll
        for (int g = 0; g < 4; ++g) G[g] = wmma_f16(Ah[kc].v, Bh[g], G[g]);
      }
      const float bI = bg[0 * 128 + ft * 16 + row];
      const float bF = bg[1 * 128 + ft * 16 + row];
      const float bG = bg[2 * 128 + ft * 16 + row];
      const float bO = bg[3 * 128 + ft * 16 + row];
      #pragma unroll
      for (int r = 0; r < 8; ++r) {       // C-layout: lane=col, vgpr=row
        float iv = sigm(G[0][r] + bI);
        float fv = sigm(G[1][r] + bF);
        float gv = tanh_s(G[2][r] + bG);
        float ov = sigm(G[3][r] + bO);
        float cv = fv * c[ft][r] + iv * gv;
        c[ft][r] = cv;
        float hv = ov * tanh_s(cv);
        hlds[(r + hi8) * DP + ft * 16 + row] = (h16)hv;   // transpose via LDS
      }
    }
    // Reload h as A-fragment for next step (same-wave LDS is in-order).
    #pragma unroll
    for (int kc = 0; kc < 4; ++kc) {
      Ah[kc].h[0] = *(const v8h*)(hlds + row * DP + kc * 32 + hi8 * 2);
      Ah[kc].h[1] = *(const v8h*)(hlds + row * DP + kc * 32 + hi8 * 2 + 8);
    }
  }

  // Output projection: h @ Wself.T + h_neigh @ Wneigh.T + b
  const h16* srow = hin + (size_t)node * DP;
  AFrag As[4];
  #pragma unroll
  for (int kc = 0; kc < 4; ++kc) {
    As[kc].h[0] = *(const v8h*)(srow + kc * 32 + hi8 * 2);
    As[kc].h[1] = *(const v8h*)(srow + kc * 32 + hi8 * 2 + 8);
  }
  const h16* wsf = (const h16*)(wblk + OFF_WSF);
  const h16* wng = (const h16*)(wblk + OFF_WNG);
  const float* bo = (const float*)(wblk + OFF_BO);
  const int ntOut = (mode == 1) ? 1 : 8;  // final layer: only col 0 is real

  #pragma unroll 1
  for (int ot = 0; ot < ntOut; ++ot) {
    v8f Cs = zz, Cn = zz;                 // independent accumulators
    #pragma unroll
    for (int kc = 0; kc < 4; ++kc) {
      v16h Bs = *(const v16h*)(wsf + ((ot * 4 + kc) * 32 + lane) * 16);
      v16h Bn = *(const v16h*)(wng + ((ot * 4 + kc) * 32 + lane) * 16);
      Cs = wmma_f16(As[kc].v, Bs, Cs);
      Cn = wmma_f16(Ah[kc].v, Bn, Cn);
    }
    v8f Cc = Cs + Cn;
    const float bb = bo[ot * 16 + row];
    if (mode == 1) {
      if (row == 0) {                     // column 0 lives in lanes 0 and 16
        #pragma unroll
        for (int r = 0; r < 8; ++r) {
          float vv = Cc[r] + bb;
          dfinal[tile * 16 + r + hi8] = 1.0f / (1.0f + __expf(-vv));
        }
      }
    } else {
      #pragma unroll
      for (int r = 0; r < 8; ++r) {
        float vv = Cc[r] + bb;
        vv = vv > 0.0f ? vv : 0.0f;       // relu between layers
        hout[(size_t)(tile * 16 + r + hi8) * DP + ot * 16 + row] = (h16)vv;
      }
    }
  }
}

// ---------------------------------------------------------------------------
// Host orchestration. ws layout:
//   [0, 5.12MB)      hbuf0 (f16, N x 128)
//   [5.12, 10.24MB)  hbuf1
//   [10.24MB, ...)   4 x LBLOCK packed weight blocks  (total ~11.56 MB)
// ---------------------------------------------------------------------------
extern "C" void kernel_launch(void* const* d_in, const int* in_sizes, int n_in,
                              void* d_out, int out_size, void* d_ws,
                              size_t ws_size, hipStream_t stream) {
  (void)in_sizes; (void)n_in; (void)out_size; (void)ws_size;
  const float* x = (const float*)d_in[0];
  const int* nbr = (const int*)d_in[1];
  char* ws = (char*)d_ws;
  h16* hb[2] = { (h16*)ws, (h16*)(ws + (size_t)N_NODES * DP * 2) };
  char* wbase = ws + 2 * (size_t)N_NODES * DP * 2;

  static const int DIN[4]  = {128, 128, 100, 128};
  static const int DOUT[4] = {128, 100, 128, 1};

  hipFuncSetAttribute(reinterpret_cast<const void*>(sage_layer),
                      hipFuncAttributeMaxDynamicSharedMemorySize, SMEM_BYTES);

  for (int l = 0; l < 4; ++l) {
    const float* Wih    = (const float*)d_in[2 + 7 * l + 0];
    const float* Whh    = (const float*)d_in[2 + 7 * l + 1];
    const float* bih    = (const float*)d_in[2 + 7 * l + 2];
    const float* bhh    = (const float*)d_in[2 + 7 * l + 3];
    const float* Wself  = (const float*)d_in[2 + 7 * l + 4];
    const float* Wneigh = (const float*)d_in[2 + 7 * l + 5];
    const float* bvec   = (const float*)d_in[2 + 7 * l + 6];
    prep_weights<<<(164480 + 255) / 256, 256, 0, stream>>>(
        Wih, Whh, bih, bhh, Wself, Wneigh, bvec, DIN[l], DOUT[l],
        wbase + (size_t)l * LBLOCK);
  }
  prep_x<<<(N_NODES * DP + 255) / 256, 256, 0, stream>>>(x, hb[0],
                                                         N_NODES * DP);

  for (int l = 0; l < 4; ++l) {
    const h16* hin = hb[l & 1];
    h16* houtp = hb[(l + 1) & 1];
    int mode = (l == 3) ? 1 : 0;
    sage_layer<<<(NTILES + WPB - 1) / WPB, 256, SMEM_BYTES, stream>>>(
        hin, nbr, wbase + (size_t)l * LBLOCK, mode ? nullptr : houtp,
        mode ? (float*)d_out : nullptr, mode);
  }
}
